// SelfAttention_28999619182818
// MI455X (gfx1250) — compile-verified, matrix operand order checked by reading, not run on hardware
//
#include <hip/hip_runtime.h>

typedef __bf16 bf16_t;
typedef __attribute__((ext_vector_type(16))) __bf16 v16bf;
typedef __attribute__((ext_vector_type(8)))  __bf16 v8bf;
typedef __attribute__((ext_vector_type(8)))  float  v8f;

#define S_LEN 4096
#define E_DIM 512
#define HD    64

// 1/sqrt(64) * log2(e): fold softmax scaling into the exp2 domain
#define SCALE_LOG2E 0.1803368801111204f

// ---------------------------------------------------------------------------
// WMMA helper: D = A(16x32 bf16) * B(32x16 bf16) + C(16x16 f32)
// ---------------------------------------------------------------------------
__device__ __forceinline__ v8f wmma_bf16(v16bf a, v16bf b, v8f c) {
    return __builtin_amdgcn_wmma_f32_16x16x32_bf16(
        /*neg_a=*/false, a, /*neg_b=*/false, b,
        /*c_mod=*/(short)0, c, /*reuse_a=*/false, /*reuse_b=*/false);
}

__device__ __forceinline__ v8f zero_v8f() {
    return (v8f){0.f,0.f,0.f,0.f,0.f,0.f,0.f,0.f};
}

// ---------------------------------------------------------------------------
// Load a 16x32 bf16 fragment in the CDNA5 A-matrix layout from a row-major
// matrix (also used for B of an N-T GEMM, loading from the row-major B^T).
// Lane l (0..15): row = l, K = {0..7, 16..23}; lane l+16: row = l, K = {8..15, 24..31}
// ---------------------------------------------------------------------------
__device__ __forceinline__ v16bf load_frag_b16(const bf16_t* base, int ld) {
    int lane = threadIdx.x & 31;
    const bf16_t* p = base + (size_t)(lane & 15) * ld + ((lane >> 4) << 3);
    v8bf lo = *(const v8bf*)(p);
    v8bf hi = *(const v8bf*)(p + 16);
    v16bf f;
#pragma unroll
    for (int i = 0; i < 8; ++i) { f[i] = lo[i]; f[i + 8] = hi[i]; }
    return f;
}

// Same fragment pattern, but source is fp32 (converted to bf16 on the fly).
__device__ __forceinline__ v16bf load_frag_f32(const float* base, int ld) {
    int lane = threadIdx.x & 31;
    const float* p = base + (size_t)(lane & 15) * ld + ((lane >> 4) << 3);
    v16bf f;
#pragma unroll
    for (int i = 0; i < 8; ++i) {
        f[i]     = (bf16_t)p[i];
        f[i + 8] = (bf16_t)p[16 + i];
    }
    return f;
}

// ---------------------------------------------------------------------------
// Projection: out[r, h] = X[r, :] . W[h, :] + bias[h]   (nn.Linear)
// X: [B*S, 512] f32, W: [64, 512] f32 (row-major => B^T row-major => A-pattern)
// Writes bf16 either row-major [B*S, 64] or transposed Vt[B][64][S].
// One wave per 16-row block, 4 N-tiles of 16, K-loop 512/32.
// ---------------------------------------------------------------------------
__launch_bounds__(128)
__global__ void proj_kernel(const float* __restrict__ X,
                            const float* __restrict__ W,
                            const float* __restrict__ bias,
                            bf16_t* __restrict__ out_rm,
                            bf16_t* __restrict__ out_tr) {
    int wave = threadIdx.x >> 5;
    int lane = threadIdx.x & 31;
    int gw   = blockIdx.x * 4 + wave;   // 0..1023
    int r0   = gw << 4;                 // global row start in [0, B*S)

    const float* Xb = X + (size_t)r0 * E_DIM;

    v8f acc[4];
#pragma unroll
    for (int t = 0; t < 4; ++t) acc[t] = zero_v8f();

    for (int kk = 0; kk < E_DIM; kk += 32) {
        v16bf a = load_frag_f32(Xb + kk, E_DIM);
#pragma unroll
        for (int t = 0; t < 4; ++t) {
            v16bf bw = load_frag_f32(W + (size_t)(t * 16) * E_DIM + kk, E_DIM);
            acc[t] = wmma_bf16(a, bw, acc[t]);
        }
    }

    int half = (lane >> 4) << 3;        // 0 or 8: row offset of this lane group
#pragma unroll
    for (int t = 0; t < 4; ++t) {
        int   h  = t * 16 + (lane & 15);
        float bb = bias[h];
#pragma unroll
        for (int v = 0; v < 8; ++v) {
            float val = acc[t][v] + bb;
            int r = r0 + v + half;
            if (out_rm) out_rm[(size_t)r * HD + h] = (bf16_t)val;
            if (out_tr) {
                int bat = r >> 12;       // r / 4096
                int s   = r & (S_LEN - 1);
                out_tr[((size_t)bat * HD + h) * S_LEN + s] = (bf16_t)val;
            }
        }
    }
}

// ---------------------------------------------------------------------------
// Fused flash attention: one wave per 16-query block, streams 4096 keys in
// 64-key chunks. Online softmax in the exp2 domain, f32 accumulators.
//   scores = Q K^T * c : 8 WMMAs/chunk (N-T GEMM, both operands A-pattern)
//   O += P V           : 8 WMMAs/chunk (via Vt so it is also N-T)
// P goes C-layout -> LDS -> A-layout (per-wave 2KB tile, in-wave DS ordering).
// Row sums kept as lane-partials (rescale factor is uniform per 16-lane
// group), reduced across lanes only in the epilogue.
// ---------------------------------------------------------------------------
__launch_bounds__(128)
__global__ void attn_kernel(const bf16_t* __restrict__ Qb,
                            const bf16_t* __restrict__ Kb,
                            const bf16_t* __restrict__ Vt,
                            float* __restrict__ out) {
    __shared__ bf16_t ptile[4][16 * 64];

    int wave = threadIdx.x >> 5;
    int lane = threadIdx.x & 31;
    int gw   = blockIdx.x * 4 + wave;   // 0..1023
    int bat  = gw >> 8;                 // 256 q-blocks per batch
    int q0   = (gw & 255) << 4;

    const bf16_t* Qbase = Qb + ((size_t)bat * S_LEN + q0) * HD;
    const bf16_t* Kbase = Kb + (size_t)bat * S_LEN * HD;
    const bf16_t* Vbase = Vt + (size_t)bat * HD * S_LEN;

    v16bf aq0 = load_frag_b16(Qbase, HD);       // Q[:, 0:32]
    v16bf aq1 = load_frag_b16(Qbase + 32, HD);  // Q[:, 32:64]

    v8f acc[4];
#pragma unroll
    for (int t = 0; t < 4; ++t) acc[t] = zero_v8f();
    float m[8], l[8];
#pragma unroll
    for (int v = 0; v < 8; ++v) { m[v] = -1e30f; l[v] = 0.f; }

    bf16_t* pt   = ptile[wave];
    int     half = (lane >> 4) << 3;    // row offset of this lane group
    int     col  = lane & 15;

    for (int kb = 0; kb < S_LEN; kb += 64) {
        // prefetch next K / V chunks (emits global_prefetch_b8)
        if (kb + 64 < S_LEN) {
            __builtin_prefetch(Kbase + (size_t)(kb + 64) * HD + lane * 4, 0, 3);
            __builtin_prefetch(Vbase + (size_t)col * S_LEN + kb + 64, 0, 3);
        }

        // --- scores: 4 tiles of 16 keys, K-dim = 64 (2 WMMAs each) ---
        v8f s[4];
#pragma unroll
        for (int t = 0; t < 4; ++t) {
            const bf16_t* kp = Kbase + (size_t)(kb + 16 * t) * HD;
            v16bf b0 = load_frag_b16(kp, HD);
            v16bf b1 = load_frag_b16(kp + 32, HD);
            v8f z = zero_v8f();
            z = wmma_bf16(aq0, b0, z);
            z = wmma_bf16(aq1, b1, z);
            s[t] = z;
        }

        // --- online softmax (exp2 domain), rescale, write P tile to LDS ---
#pragma unroll
        for (int v = 0; v < 8; ++v) {
            float a0 = s[0][v] * SCALE_LOG2E;
            float a1 = s[1][v] * SCALE_LOG2E;
            float a2 = s[2][v] * SCALE_LOG2E;
            float a3 = s[3][v] * SCALE_LOG2E;
            float mx = fmaxf(fmaxf(a0, a1), fmaxf(a2, a3));
            mx = fmaxf(mx, __shfl_xor(mx, 1, 32));
            mx = fmaxf(mx, __shfl_xor(mx, 2, 32));
            mx = fmaxf(mx, __shfl_xor(mx, 4, 32));
            mx = fmaxf(mx, __shfl_xor(mx, 8, 32));
            float mn = fmaxf(m[v], mx);
            float sc = exp2f(m[v] - mn);
            m[v] = mn;
            float p0 = exp2f(a0 - mn);
            float p1 = exp2f(a1 - mn);
            float p2 = exp2f(a2 - mn);
            float p3 = exp2f(a3 - mn);
            // lane-partial row sum; sc is uniform within the 16-lane group
            l[v] = l[v] * sc + ((p0 + p1) + (p2 + p3));
            acc[0][v] *= sc; acc[1][v] *= sc;
            acc[2][v] *= sc; acc[3][v] *= sc;
            // C-layout element (row = v+half, key col) -> row-major LDS P tile
            int r = v + half;
            pt[r * 64 + col]      = (bf16_t)p0;
            pt[r * 64 + 16 + col] = (bf16_t)p1;
            pt[r * 64 + 32 + col] = (bf16_t)p2;
            pt[r * 64 + 48 + col] = (bf16_t)p3;
        }

        // --- O += P V : P back in A-layout (2 frags), 4 H-tiles ---
        v16bf ap0 = load_frag_b16(pt, 64);       // P[:, 0:32]
        v16bf ap1 = load_frag_b16(pt + 32, 64);  // P[:, 32:64]
#pragma unroll
        for (int t = 0; t < 4; ++t) {
            // B = V[kb:kb+64, 16t:16t+16]; B^T rows are Vt rows -> A-pattern
            const bf16_t* vp = Vbase + (size_t)(t * 16) * S_LEN + kb;
            acc[t] = wmma_bf16(ap0, load_frag_b16(vp, S_LEN), acc[t]);
            acc[t] = wmma_bf16(ap1, load_frag_b16(vp + 32, S_LEN), acc[t]);
        }
    }

    // epilogue: reduce lane-partial row sums once, O / l, store f32 [B, S, 64]
#pragma unroll
    for (int v = 0; v < 8; ++v) {
        float lv = l[v];
        lv += __shfl_xor(lv, 1, 32);
        lv += __shfl_xor(lv, 2, 32);
        lv += __shfl_xor(lv, 4, 32);
        lv += __shfl_xor(lv, 8, 32);
        float inv = 1.0f / lv;
        int r = q0 + v + half;
        float* orow = out + ((size_t)bat * S_LEN + r) * HD + col;
        orow[0]  = acc[0][v] * inv;
        orow[16] = acc[1][v] * inv;
        orow[32] = acc[2][v] * inv;
        orow[48] = acc[3][v] * inv;
    }
}

// ---------------------------------------------------------------------------
extern "C" void kernel_launch(void* const* d_in, const int* in_sizes, int n_in,
                              void* d_out, int out_size, void* d_ws, size_t ws_size,
                              hipStream_t stream) {
    const float* query = (const float*)d_in[0];
    const float* key_  = (const float*)d_in[1];
    const float* value = (const float*)d_in[2];
    const float* Wq    = (const float*)d_in[3];
    const float* bq    = (const float*)d_in[4];
    const float* Wk    = (const float*)d_in[5];
    const float* bk    = (const float*)d_in[6];
    const float* Wv    = (const float*)d_in[7];
    const float* bv    = (const float*)d_in[8];

    // workspace: Qb (2MB) | Kb (2MB) | Vt (2MB), all bf16
    const size_t n_qkv = (size_t)4 * S_LEN * HD;
    bf16_t* Qb = (bf16_t*)d_ws;
    bf16_t* Kb = Qb + n_qkv;
    bf16_t* Vt = Kb + n_qkv;

    dim3 blk(128);          // 4 waves (wave32)
    dim3 grd(256);          // 1024 waves = (B*S)/16 row-blocks

    proj_kernel<<<grd, blk, 0, stream>>>(query, Wq, bq, Qb, nullptr);
    proj_kernel<<<grd, blk, 0, stream>>>(key_,  Wk, bk, Kb, nullptr);
    proj_kernel<<<grd, blk, 0, stream>>>(value, Wv, bv, nullptr, Vt);
    attn_kernel<<<grd, blk, 0, stream>>>(Qb, Kb, Vt, (float*)d_out);
}
